// HGTModel_24154896073601
// MI455X (gfx1250) — matrix operand order repaired and checked by reading, not compiled, not run on hardware
//
#include <hip/hip_runtime.h>
#include <math.h>

#define HID 128
#define T   4
#define R   5
#define H   4
#define L   2
#define DK  32

typedef __attribute__((ext_vector_type(16))) __bf16 v16bf;
typedef __attribute__((ext_vector_type(8)))  __bf16 v8bf;
typedef __attribute__((ext_vector_type(8)))  float  v8f;

__device__ __forceinline__ __bf16 f2bf(float f) {
  unsigned u = __float_as_uint(f);
  u += 0x7FFFu + ((u >> 16) & 1u);        // round-to-nearest-even
  unsigned short hs = (unsigned short)(u >> 16);
  return __builtin_bit_cast(__bf16, hs);
}

__device__ __forceinline__ float geluf(float x) {
  float x3 = x * x * x;
  return 0.5f * x * (1.0f + tanhf(0.7978845608028654f * (x + 0.044715f * x3)));
}

__device__ __forceinline__ void atomicMaxF(float* addr, float val) {
  // valid because buffer is initialized to -inf and values are finite
  if (val >= 0.0f) atomicMax((int*)addr, __float_as_int(val));
  else             atomicMin((unsigned int*)addr, __float_as_uint(val));
}

// ---------------------------------------------------------------------------
// Per-node-type linear via WMMA bf16:
//   out_row = x_row @ W[type(row)] + b[type(row)]
// computed as sum over t of (mask_t(x) @ W[t]) in one WMMA accumulator chain.
// MODE 0: post tanh (adapt). MODE 1: plain (k/q/v). MODE 2: pre-gelu on X,
// post gated-residual blend with Hin using sigmoid(Skip[type]).
// Block = 128 threads (4 waves); tile = 16 rows x 128 cols; each wave owns
// two 16x16 output tiles. Wt is bf16, transposed: [T][n=128][k=128].
// ---------------------------------------------------------------------------
template <int MODE>
__global__ __launch_bounds__(128)
void typelinear_kernel(const float* __restrict__ X,
                       const int*   __restrict__ ntype,
                       const __bf16* __restrict__ Wt,
                       const float* __restrict__ Bias,   // [T,128]
                       const float* __restrict__ Hin,    // MODE 2 only
                       const float* __restrict__ Skip,   // MODE 2 only, [T]
                       float* __restrict__ Out,
                       int nrows)
{
  __shared__ __bf16 sX[16][HID + 8];    // +8 bf16 pad -> conflict-free b128 reads
  __shared__ int    sType[16];
  __shared__ int    sHas[T];

  const int m0  = blockIdx.x * 16;
  const int tid = threadIdx.x;

  // Stage tile, converting to bf16 once (pre-activation applied here).
  for (int idx = tid; idx < 16 * HID; idx += 128) {
    int r = idx >> 7, c = idx & 127;
    int row = m0 + r;
    float v = (row < nrows) ? X[(size_t)row * HID + c] : 0.0f;
    if (MODE == 2) v = geluf(v);
    sX[r][c] = f2bf(v);
  }
  if (tid < 16) {
    int row = m0 + tid;
    sType[tid] = (row < nrows) ? ntype[row] : -1;
  }
  __syncthreads();
  if (tid < T) {
    int f = 0;
    for (int r = 0; r < 16; ++r) f |= (sType[r] == tid) ? 1 : 0;
    sHas[tid] = f;
  }
  __syncthreads();

  const int lane   = tid & 31;
  const int wave   = tid >> 5;
  const int laneLo = lane & 15;
  const int laneHi = lane >> 4;
  const int n0     = wave * 32;
  const int ar     = laneLo;                 // A-matrix row owned by this lane

  // Flags / row type into registers (one-time LDS reads).
  int hasMask = 0;
  #pragma unroll
  for (int t = 0; t < T; ++t) hasMask |= (sHas[t] ? 1 : 0) << t;
  const int myType = sType[ar];

  // Hoist all four K-slice A fragments out of the type loop.
  // A fragment layout (16x32 bf16): lane<16 -> K{0..15}, lane>=16 -> K{16..31}
  v16bf afrag[4];
  #pragma unroll
  for (int kk = 0; kk < 4; ++kk) {
    const int kb = kk * 32 + laneHi * 16;
    v8bf lo = *reinterpret_cast<const v8bf*>(&sX[ar][kb]);
    v8bf hi = *reinterpret_cast<const v8bf*>(&sX[ar][kb + 8]);
    afrag[kk] = __builtin_shufflevector(lo, hi, 0, 1, 2, 3, 4, 5, 6, 7,
                                        8, 9, 10, 11, 12, 13, 14, 15);
  }
  const v16bf az = {};

  v8f c0 = {}; v8f c1 = {};

  for (int t = 0; t < T; ++t) {
    if (!((hasMask >> t) & 1)) continue;     // uniform branch (EXEC stays full)
    const bool am = (myType == t);
    #pragma unroll
    for (int kk = 0; kk < 4; ++kk) {
      v16bf a = am ? afrag[kk] : az;         // single vector select, no loads
      // B fragment: 32x16 bf16; Wt transposed so each lane reads 16 contiguous bf16
      const int khalf = kk * 32 + laneHi * 16;
      const __bf16* w0 = Wt + ((size_t)(t * HID + n0 + laneLo)      * HID + khalf);
      const __bf16* w1 = Wt + ((size_t)(t * HID + n0 + 16 + laneLo) * HID + khalf);
      v16bf b0 = *reinterpret_cast<const v16bf*>(w0);
      v16bf b1 = *reinterpret_cast<const v16bf*>(w1);
      c0 = __builtin_amdgcn_wmma_f32_16x16x32_bf16(false, a, false, b0, (short)0, c0, false, false);
      c1 = __builtin_amdgcn_wmma_f32_16x16x32_bf16(false, a, false, b1, (short)0, c1, false, false);
    }
  }

  // C layout: VGPR v -> row (v + laneHi*8), col = n0 + laneLo (+16 for c1)
  #pragma unroll
  for (int v = 0; v < 8; ++v) {
    int rr  = v + laneHi * 8;
    int row = m0 + rr;
    if (row >= nrows) continue;
    int tt   = sType[rr];
    int col0 = n0 + laneLo;
    int col1 = col0 + 16;
    float o0 = c0[v] + Bias[tt * HID + col0];
    float o1 = c1[v] + Bias[tt * HID + col1];
    if (MODE == 0) { o0 = tanhf(o0); o1 = tanhf(o1); }
    if (MODE == 2) {
      float sk = 1.0f / (1.0f + __expf(-Skip[tt]));
      o0 = o0 * sk + Hin[(size_t)row * HID + col0] * (1.0f - sk);
      o1 = o1 * sk + Hin[(size_t)row * HID + col1] * (1.0f - sk);
    }
    Out[(size_t)row * HID + col0] = o0;
    Out[(size_t)row * HID + col1] = o1;
  }
}

// Convert W [nmats][k=128][n=128] f32 -> Wt [nmats][n=128][k=128] bf16
__global__ void wconv_kernel(const float* __restrict__ W, __bf16* __restrict__ Wt, int nmats) {
  int i = blockIdx.x * blockDim.x + threadIdx.x;
  if (i >= (nmats << 14)) return;
  int m = i >> 14, rem = i & 16383;
  int n = rem >> 7, k = rem & 127;
  Wt[i] = f2bf(W[(m << 14) + (k << 7) + n]);
}

__global__ void fill_kernel(float* __restrict__ p, float v, int n) {
  int i = blockIdx.x * blockDim.x + threadIdx.x;
  if (i < n) p[i] = v;
}

// Pass 1: per (edge, head): att = (q[dst] . (k[src] @ rel_att[r])) * pri / sqrt(DK)
// One wave per (edge, head); relation matrices for this head staged in LDS.
__global__ __launch_bounds__(256)
void edge_att_kernel(const float* __restrict__ kbuf, const float* __restrict__ qbuf,
                     const int* __restrict__ src, const int* __restrict__ dst,
                     const int* __restrict__ etype,
                     const float* __restrict__ rel_att,   // [R,H,DK,DK]
                     const float* __restrict__ rel_pri,   // [R,H]
                     float* __restrict__ att,             // [E,H]
                     float* __restrict__ amax,            // [N,H]
                     int nedges)
{
  const int h = blockIdx.y;
  __shared__ float sA[R * DK * DK];
  __shared__ float sPri[R];
  for (int i = threadIdx.x; i < R * DK * DK; i += blockDim.x) {
    int r = i >> 10, rem = i & 1023;
    sA[i] = rel_att[((r * H + h) << 10) + rem];
  }
  if (threadIdx.x < R) sPri[threadIdx.x] = rel_pri[threadIdx.x * H + h];
  __syncthreads();

  const int lane = threadIdx.x & 31;
  const int wpb  = blockDim.x >> 5;
  const int nw   = gridDim.x * wpb;
  const float inv_sqrt_dk = 0.17677669529663687f;

  for (int e = blockIdx.x * wpb + (threadIdx.x >> 5); e < nedges; e += nw) {
    int s = src[e], d = dst[e], r = etype[e];
    float kev = kbuf[(size_t)s * HID + h * DK + lane];
    float qev = qbuf[(size_t)d * HID + h * DK + lane];
    const float* A = sA + (r << 10);
    float kt = 0.0f;
    #pragma unroll
    for (int i = 0; i < DK; ++i) {
      float ki = __shfl(kev, i, 32);           // uniform lane -> readlane broadcast
      kt = fmaf(ki, A[(i << 5) + lane], kt);
    }
    float p = qev * kt;
    #pragma unroll
    for (int off = 16; off; off >>= 1) p += __shfl_xor(p, off, 32);
    if (lane == 0) {
      float a = p * sPri[r] * inv_sqrt_dk;
      att[(size_t)e * H + h] = a;
      atomicMaxF(&amax[d * H + h], a);
    }
  }
}

// Pass 2: ex = exp(att - amax[dst]); den[dst] += ex  (att overwritten with ex)
__global__ void edge_softmax_kernel(const int* __restrict__ dst,
                                    float* __restrict__ att,
                                    const float* __restrict__ amax,
                                    float* __restrict__ den,
                                    int nEH)
{
  int i = blockIdx.x * blockDim.x + threadIdx.x;
  if (i >= nEH) return;
  int e = i >> 2, h = i & 3;                   // H == 4
  int d = dst[e];
  float ex = __expf(att[i] - amax[d * H + h]);
  att[i] = ex;
  atomicAdd(&den[d * H + h], ex);
}

// Pass 3: res[dst] += alpha * (v[src] @ rel_msg[r])
__global__ __launch_bounds__(256)
void edge_msg_kernel(const float* __restrict__ vbuf,
                     const int* __restrict__ src, const int* __restrict__ dst,
                     const int* __restrict__ etype,
                     const float* __restrict__ rel_msg,   // [R,H,DK,DK]
                     const float* __restrict__ ex,        // [E,H]
                     const float* __restrict__ den,       // [N,H]
                     float* __restrict__ res,             // [N,HID]
                     int nedges)
{
  const int h = blockIdx.y;
  __shared__ float sM[R * DK * DK];
  for (int i = threadIdx.x; i < R * DK * DK; i += blockDim.x) {
    int r = i >> 10, rem = i & 1023;
    sM[i] = rel_msg[((r * H + h) << 10) + rem];
  }
  __syncthreads();

  const int lane = threadIdx.x & 31;
  const int wpb  = blockDim.x >> 5;
  const int nw   = gridDim.x * wpb;

  for (int e = blockIdx.x * wpb + (threadIdx.x >> 5); e < nedges; e += nw) {
    int s = src[e], d = dst[e], r = etype[e];
    float vev = vbuf[(size_t)s * HID + h * DK + lane];
    const float* M = sM + (r << 10);
    float mt = 0.0f;
    #pragma unroll
    for (int i = 0; i < DK; ++i) {
      float vi = __shfl(vev, i, 32);
      mt = fmaf(vi, M[(i << 5) + lane], mt);
    }
    float alpha = ex[(size_t)e * H + h] / fmaxf(den[d * H + h], 1e-9f);
    atomicAdd(&res[(size_t)d * HID + h * DK + lane], alpha * mt);
  }
}

extern "C" void kernel_launch(void* const* d_in, const int* in_sizes, int n_in,
                              void* d_out, int out_size, void* d_ws, size_t ws_size,
                              hipStream_t stream)
{
  const float* node_feature = (const float*)d_in[0];
  const int*   node_type    = (const int*)d_in[1];
  // d_in[2] edge_time unused by the reference
  const int*   edge_index   = (const int*)d_in[3];
  const int*   edge_type    = (const int*)d_in[4];
  const float* adapt_w = (const float*)d_in[5];
  const float* adapt_b = (const float*)d_in[6];
  const float* kw = (const float*)d_in[7];
  const float* kb = (const float*)d_in[8];
  const float* qw = (const float*)d_in[9];
  const float* qb = (const float*)d_in[10];
  const float* vw = (const float*)d_in[11];
  const float* vb = (const float*)d_in[12];
  const float* aw = (const float*)d_in[13];
  const float* ab = (const float*)d_in[14];
  const float* rel_pri = (const float*)d_in[15];
  const float* rel_att = (const float*)d_in[16];
  const float* rel_msg = (const float*)d_in[17];
  const float* skip    = (const float*)d_in[18];

  const int N = in_sizes[1];
  const int E = in_sizes[4];
  const int* src = edge_index;
  const int* dst = edge_index + E;

  char* ws = (char*)d_ws;
  size_t off = 0;
  auto alloc = [&](size_t bytes) -> void* {
    void* p = ws + off;
    off = (off + bytes + 255) & ~(size_t)255;
    return p;
  };
  float*  hA     = (float*)alloc((size_t)N * HID * 4);
  float*  hB     = (float*)alloc((size_t)N * HID * 4);
  float*  kbuf   = (float*)alloc((size_t)N * HID * 4);
  float*  qbuf   = (float*)alloc((size_t)N * HID * 4);
  float*  vbuf   = (float*)alloc((size_t)N * HID * 4);
  float*  res    = (float*)alloc((size_t)N * HID * 4);
  float*  attb   = (float*)alloc((size_t)E * H * 4);
  float*  amax   = (float*)alloc((size_t)N * H * 4);
  float*  den    = (float*)alloc((size_t)N * H * 4);
  __bf16* adaptT = (__bf16*)alloc((size_t)T * HID * HID * 2);
  __bf16* kwT    = (__bf16*)alloc((size_t)L * T * HID * HID * 2);
  __bf16* qwT    = (__bf16*)alloc((size_t)L * T * HID * HID * 2);
  __bf16* vwT    = (__bf16*)alloc((size_t)L * T * HID * HID * 2);
  __bf16* awT    = (__bf16*)alloc((size_t)L * T * HID * HID * 2);
  (void)n_in; (void)out_size; (void)ws_size;

  // 1) transpose+convert weights to bf16 (tiny; L2-resident thereafter)
  {
    int nm = T;
    wconv_kernel<<<(nm * 16384 + 255) / 256, 256, 0, stream>>>(adapt_w, adaptT, nm);
    nm = L * T;
    wconv_kernel<<<(nm * 16384 + 255) / 256, 256, 0, stream>>>(kw, kwT, nm);
    wconv_kernel<<<(nm * 16384 + 255) / 256, 256, 0, stream>>>(qw, qwT, nm);
    wconv_kernel<<<(nm * 16384 + 255) / 256, 256, 0, stream>>>(vw, vwT, nm);
    wconv_kernel<<<(nm * 16384 + 255) / 256, 256, 0, stream>>>(aw, awT, nm);
  }

  const int gemmBlocks = (N + 15) / 16;

  // 2) per-type adapt + tanh
  typelinear_kernel<0><<<gemmBlocks, 128, 0, stream>>>(
      node_feature, node_type, adaptT, adapt_b, nullptr, nullptr, hA, N);

  dim3 egrid(1024, H);
  for (int l = 0; l < L; ++l) {
    const size_t wmoff = (size_t)l * T * HID * HID;

    typelinear_kernel<1><<<gemmBlocks, 128, 0, stream>>>(
        hA, node_type, kwT + wmoff, kb + (size_t)l * T * HID, nullptr, nullptr, kbuf, N);
    typelinear_kernel<1><<<gemmBlocks, 128, 0, stream>>>(
        hA, node_type, qwT + wmoff, qb + (size_t)l * T * HID, nullptr, nullptr, qbuf, N);
    typelinear_kernel<1><<<gemmBlocks, 128, 0, stream>>>(
        hA, node_type, vwT + wmoff, vb + (size_t)l * T * HID, nullptr, nullptr, vbuf, N);

    fill_kernel<<<(N * H + 255) / 256, 256, 0, stream>>>(amax, -INFINITY, N * H);
    fill_kernel<<<(N * H + 255) / 256, 256, 0, stream>>>(den, 0.0f, N * H);
    fill_kernel<<<(N * HID + 255) / 256, 256, 0, stream>>>(res, 0.0f, N * HID);

    const float* ratt = rel_att + (size_t)l * R * H * DK * DK;
    const float* rmsg = rel_msg + (size_t)l * R * H * DK * DK;
    const float* rpri = rel_pri + (size_t)l * R * H;

    edge_att_kernel<<<egrid, 256, 0, stream>>>(kbuf, qbuf, src, dst, edge_type,
                                               ratt, rpri, attb, amax, E);
    edge_softmax_kernel<<<(E * H + 255) / 256, 256, 0, stream>>>(dst, attb, amax, den, E * H);
    edge_msg_kernel<<<egrid, 256, 0, stream>>>(vbuf, src, dst, edge_type,
                                               rmsg, attb, den, res, E);

    float* outp = (l == L - 1) ? (float*)d_out : hB;
    typelinear_kernel<2><<<gemmBlocks, 128, 0, stream>>>(
        res, node_type, awT + wmoff, ab + (size_t)l * T * HID,
        hA, skip + (size_t)l * T, outp, N);

    float* tmp = hA; hA = hB; hB = tmp;
  }
}
